// PoseNetX_LIGHT_82772609729157
// MI455X (gfx1250) — matrix operand order repaired and checked by reading, not compiled, not run on hardware
//
#include <hip/hip_runtime.h>
#include <hip/hip_bf16.h>

typedef __attribute__((ext_vector_type(16))) __bf16 v16bf;
typedef __attribute__((ext_vector_type(8)))  float  v8f;
typedef unsigned short ushort_t;

#define N_NODES 256
#define N_EDGES 8192
#define IMG     256
#define FEAT    2048
#define EFEAT   2048
#define PLANES  512
#define KCONV   160   // 3*7*7 = 147 padded to multiple of 32

union FragAB { v16bf v; uint4 q[2]; };

__device__ __forceinline__ ushort_t f32_to_bf16(float f) {
    unsigned int u = __float_as_uint(f);
    u = u + 0x7FFFu + ((u >> 16) & 1u);   // round-to-nearest-even
    return (ushort_t)(u >> 16);
}

// ---------------- conversion / packing kernels ----------------

__global__ void k_zero_f32(float* p, int n) {
    int i = blockIdx.x * blockDim.x + threadIdx.x;
    if (i < n) p[i] = 0.0f;
}

__global__ void k_f32_to_bf16(const float* __restrict__ s, ushort_t* __restrict__ d, int n) {
    int i = blockIdx.x * blockDim.x + threadIdx.x;
    if (i < n) d[i] = f32_to_bf16(s[i]);
}

// conv_w [512,3,7,7] f32  ->  [512, 160] bf16 (k padded with zeros)
__global__ void k_pack_convw(const float* __restrict__ w, ushort_t* __restrict__ d) {
    int i = blockIdx.x * blockDim.x + threadIdx.x;
    if (i >= PLANES * KCONV) return;
    int oc = i / KCONV, k = i % KCONV;
    d[i] = (k < 147) ? f32_to_bf16(w[oc * 147 + k]) : (ushort_t)0;
}

// proj_w [2048, 4096] f32 -> W1 [2048,2048] bf16, W2 [2048,2048] bf16
__global__ void k_split_projw(const float* __restrict__ pw,
                              ushort_t* __restrict__ w1, ushort_t* __restrict__ w2) {
    int i = blockIdx.x * blockDim.x + threadIdx.x;
    if (i >= EFEAT * FEAT) return;
    int n = i / FEAT, k = i % FEAT;
    w1[i] = f32_to_bf16(pw[(size_t)n * (2 * FEAT) + k]);
    w2[i] = f32_to_bf16(pw[(size_t)n * (2 * FEAT) + FEAT + k]);
}

// im2col: x [256, 3*256*256] -> A [65536, 160] bf16, row = img*256 + oy*16 + ox
__global__ void k_im2col(const float* __restrict__ x, ushort_t* __restrict__ a) {
    int i = blockIdx.x * blockDim.x + threadIdx.x;
    if (i >= 65536 * KCONV) return;
    int row = i / KCONV, k = i % KCONV;
    float v = 0.0f;
    if (k < 147) {
        int img = row >> 8, pos = row & 255;
        int oy = pos >> 4, ox = pos & 15;
        int ic = k / 49, rem = k % 49;
        int ky = rem / 7, kx = rem % 7;
        int y = oy * 16 - 3 + ky, xx = ox * 16 - 3 + kx;
        if (y >= 0 && y < IMG && xx >= 0 && xx < IMG)
            v = x[(size_t)img * (3 * IMG * IMG) + (size_t)ic * (IMG * IMG) + y * IMG + xx];
    }
    a[i] = f32_to_bf16(v);
}

// ---------------- WMMA GEMM:  C[M,N] = A[M,K] * W[N,K]^T (+bias) ----------------
// block = 256 threads = 8 waves; each wave -> 16M x 64N strip (4 independent
// accumulators, one shared A fragment per K-step); block covers 16M x 512N.
// MODE 0: store f32 C (+ optional bf16 mirror).
// MODE 1: epilogue = relu(acc + bias[n]) / 256, atomicAdd into pool[row>>8][n].
template <int MODE>
__global__ void k_gemm_bf16(const ushort_t* __restrict__ A, const ushort_t* __restrict__ B,
                            const float* __restrict__ bias,
                            float* __restrict__ C, ushort_t* __restrict__ Cbf,
                            float* __restrict__ pool, int M, int N, int K) {
    const int wave = threadIdx.x >> 5;
    const int lane = threadIdx.x & 31;
    const int hl   = lane >> 4;      // half-wave
    const int r    = lane & 15;
    const int tileM = blockIdx.y * 16;
    const int tileN = blockIdx.x * 512 + wave * 64;

    const ushort_t* pa_row = A + (size_t)(tileM + r) * K + hl * 16;
    const ushort_t* pb_row = B + (size_t)(tileN + r) * K + hl * 16;
    const size_t nstride16 = (size_t)16 * K;   // 16 columns of W

    v8f acc0 = {}, acc1 = {}, acc2 = {}, acc3 = {};
    for (int k0 = 0; k0 < K; k0 += 32) {
        FragAB fa, fb0, fb1, fb2, fb3;
        // A 16x32 bf16: lanes 0-15 K=[0..15], lanes 16-31 K=[16..31]
        const ushort_t* pa = pa_row + k0;
        fa.q[0] = *(const uint4*)(pa);
        fa.q[1] = *(const uint4*)(pa + 8);
        // B 32x16 bf16: lanes 0-15 K=[0..15], lanes 16-31 K=[16..31] of column n
        const ushort_t* pb = pb_row + k0;
        fb0.q[0] = *(const uint4*)(pb);
        fb0.q[1] = *(const uint4*)(pb + 8);
        fb1.q[0] = *(const uint4*)(pb + nstride16);
        fb1.q[1] = *(const uint4*)(pb + nstride16 + 8);
        fb2.q[0] = *(const uint4*)(pb + 2 * nstride16);
        fb2.q[1] = *(const uint4*)(pb + 2 * nstride16 + 8);
        fb3.q[0] = *(const uint4*)(pb + 3 * nstride16);
        fb3.q[1] = *(const uint4*)(pb + 3 * nstride16 + 8);
        // four independent accumulation chains -> matrix-pipe overlap
        acc0 = __builtin_amdgcn_wmma_f32_16x16x32_bf16(false, fa.v, false, fb0.v, (short)0, acc0, false, false);
        acc1 = __builtin_amdgcn_wmma_f32_16x16x32_bf16(false, fa.v, false, fb1.v, (short)0, acc1, false, false);
        acc2 = __builtin_amdgcn_wmma_f32_16x16x32_bf16(false, fa.v, false, fb2.v, (short)0, acc2, false, false);
        acc3 = __builtin_amdgcn_wmma_f32_16x16x32_bf16(false, fa.v, false, fb3.v, (short)0, acc3, false, false);
    }

    v8f accs[4] = {acc0, acc1, acc2, acc3};
    #pragma unroll
    for (int j = 0; j < 4; ++j) {
        const int col = tileN + j * 16 + r;
        const float bv = bias ? bias[col] : 0.0f;
        if (MODE == 0) {
            #pragma unroll
            for (int i = 0; i < 8; ++i) {
                int row = tileM + hl * 8 + i;
                float v = accs[j][i] + bv;
                C[(size_t)row * N + col] = v;
                if (Cbf) Cbf[(size_t)row * N + col] = f32_to_bf16(v);
            }
        } else {
            float s = 0.0f;
            #pragma unroll
            for (int i = 0; i < 8; ++i) s += fmaxf(accs[j][i] + bv, 0.0f);
            int img = tileM >> 8;   // 256 spatial positions per image, tile never straddles
            atomicAdd(&pool[(size_t)img * N + col], s * (1.0f / 256.0f));
        }
    }
}

// ---------------- per-node head: out_node[i, 0:6] ----------------
__global__ void k_node_head(const float* __restrict__ node,
                            const float* __restrict__ xw, const float* __restrict__ xb,
                            const float* __restrict__ ww, const float* __restrict__ wb,
                            float* __restrict__ out) {
    int wave = threadIdx.x >> 5, lane = threadIdx.x & 31;
    int nid = blockIdx.x * 8 + wave;
    if (nid >= N_NODES) return;
    const float* p = node + (size_t)nid * FEAT;
    float a0=0,a1=0,a2=0,a3=0,a4=0,a5=0;
    for (int j = lane; j < FEAT; j += 32) {
        float v = fmaxf(p[j], 0.0f);
        a0 += v * xw[j];        a1 += v * xw[FEAT + j];  a2 += v * xw[2*FEAT + j];
        a3 += v * ww[j];        a4 += v * ww[FEAT + j];  a5 += v * ww[2*FEAT + j];
    }
    #pragma unroll
    for (int off = 16; off > 0; off >>= 1) {
        a0 += __shfl_down(a0, off, 32); a1 += __shfl_down(a1, off, 32);
        a2 += __shfl_down(a2, off, 32); a3 += __shfl_down(a3, off, 32);
        a4 += __shfl_down(a4, off, 32); a5 += __shfl_down(a5, off, 32);
    }
    if (lane == 0) {
        float* o = out + (size_t)nid * 6;
        o[0] = a0 + xb[0]; o[1] = a1 + xb[1]; o[2] = a2 + xb[2];
        o[3] = a3 + wb[0]; o[4] = a4 + wb[1]; o[5] = a5 + wb[2];
    }
}

// ---------------- per-edge: relu(P1[n1]+P2[n2]) -> 6-dim head ----------------
// (proj_b was folded into P1's GEMM bias)
__global__ void k_edge_head(const int* __restrict__ ei,
                            const float* __restrict__ P1, const float* __restrict__ P2,
                            const float* __restrict__ xw, const float* __restrict__ xb,
                            const float* __restrict__ ww, const float* __restrict__ wb,
                            float* __restrict__ out) {
    int wave = threadIdx.x >> 5, lane = threadIdx.x & 31;
    int e = blockIdx.x * 8 + wave;
    if (e >= N_EDGES) return;
    int u = ei[e], v = ei[N_EDGES + e];
    int n1 = min(u, v), n2 = max(u, v);
    const float* p1 = P1 + (size_t)n1 * EFEAT;
    const float* p2 = P2 + (size_t)n2 * EFEAT;
    float a0=0,a1=0,a2=0,a3=0,a4=0,a5=0;
    for (int j = lane; j < EFEAT; j += 32) {
        float f = fmaxf(p1[j] + p2[j], 0.0f);
        a0 += f * xw[j];        a1 += f * xw[EFEAT + j];  a2 += f * xw[2*EFEAT + j];
        a3 += f * ww[j];        a4 += f * ww[EFEAT + j];  a5 += f * ww[2*EFEAT + j];
    }
    #pragma unroll
    for (int off = 16; off > 0; off >>= 1) {
        a0 += __shfl_down(a0, off, 32); a1 += __shfl_down(a1, off, 32);
        a2 += __shfl_down(a2, off, 32); a3 += __shfl_down(a3, off, 32);
        a4 += __shfl_down(a4, off, 32); a5 += __shfl_down(a5, off, 32);
    }
    if (lane == 0) {
        float* o = out + (size_t)e * 6;
        o[0] = a0 + xb[0]; o[1] = a1 + xb[1]; o[2] = a2 + xb[2];
        o[3] = a3 + wb[0]; o[4] = a4 + wb[1]; o[5] = a5 + wb[2];
    }
}

extern "C" void kernel_launch(void* const* d_in, const int* in_sizes, int n_in,
                              void* d_out, int out_size, void* d_ws, size_t ws_size,
                              hipStream_t stream) {
    const float* x      = (const float*)d_in[0];
    const int*   eidx   = (const int*)d_in[1];
    const float* conv_w = (const float*)d_in[2];
    const float* conv_b = (const float*)d_in[3];
    const float* fc_w   = (const float*)d_in[4];
    const float* fc_b   = (const float*)d_in[5];
    const float* proj_w = (const float*)d_in[6];
    const float* proj_b = (const float*)d_in[7];
    const float* xyz_w  = (const float*)d_in[8];
    const float* xyz_b  = (const float*)d_in[9];
    const float* wpqr_w = (const float*)d_in[10];
    const float* wpqr_b = (const float*)d_in[11];
    const float* xyzR_w = (const float*)d_in[12];
    const float* xyzR_b = (const float*)d_in[13];
    const float* wpqRw  = (const float*)d_in[14];
    const float* wpqRb  = (const float*)d_in[15];
    float* out = (float*)d_out;               // [256*6] node, then [8192*6] edge

    // ---- workspace carve (all segments 256B aligned) ----
    char* ws = (char*)d_ws;
    size_t off = 0;
    auto take = [&](size_t bytes) { void* p = ws + off; off += (bytes + 255) & ~(size_t)255; return p; };
    ushort_t* Acv   = (ushort_t*)take((size_t)65536 * KCONV * 2);   // im2col bf16
    ushort_t* Wc    = (ushort_t*)take((size_t)PLANES * KCONV * 2);  // conv weights bf16
    ushort_t* fcwB  = (ushort_t*)take((size_t)FEAT * PLANES * 2);   // fc weights bf16
    ushort_t* W1    = (ushort_t*)take((size_t)EFEAT * FEAT * 2);    // proj left half
    ushort_t* W2    = (ushort_t*)take((size_t)EFEAT * FEAT * 2);    // proj right half
    float*    pool  = (float*)   take((size_t)N_NODES * PLANES * 4);
    ushort_t* poolB = (ushort_t*)take((size_t)N_NODES * PLANES * 2);
    float*    node  = (float*)   take((size_t)N_NODES * FEAT * 4);
    ushort_t* nodeB = (ushort_t*)take((size_t)N_NODES * FEAT * 2);
    float*    P1    = (float*)   take((size_t)N_NODES * EFEAT * 4);
    float*    P2    = (float*)   take((size_t)N_NODES * EFEAT * 4);
    (void)ws_size; (void)n_in; (void)in_sizes; (void)out_size;

    const int T = 256;
    // ---- packing ----
    k_pack_convw <<<(PLANES * KCONV + T - 1) / T, T, 0, stream>>>(conv_w, Wc);
    k_f32_to_bf16<<<(FEAT * PLANES + T - 1) / T, T, 0, stream>>>(fc_w, fcwB, FEAT * PLANES);
    k_split_projw<<<(EFEAT * FEAT + T - 1) / T, T, 0, stream>>>(proj_w, W1, W2);
    k_im2col     <<<(65536 * KCONV + T - 1) / T, T, 0, stream>>>(x, Acv);
    k_zero_f32   <<<(N_NODES * PLANES + T - 1) / T, T, 0, stream>>>(pool, N_NODES * PLANES);

    // ---- conv as WMMA GEMM, fused bias+relu+avgpool epilogue ----
    k_gemm_bf16<1><<<dim3(PLANES / 512 ? PLANES / 512 : 1, 65536 / 16), T, 0, stream>>>(
        Acv, Wc, conv_b, nullptr, nullptr, pool, 65536, PLANES, KCONV);
    k_f32_to_bf16<<<(N_NODES * PLANES + T - 1) / T, T, 0, stream>>>(pool, poolB, N_NODES * PLANES);

    // ---- fc: node = pooled @ fc_w^T + fc_b  (f32 + bf16 mirror) ----
    k_gemm_bf16<0><<<dim3(FEAT / 512, N_NODES / 16), T, 0, stream>>>(
        poolB, fcwB, fc_b, node, nodeB, nullptr, N_NODES, FEAT, PLANES);

    // ---- P1 = node @ W1^T + proj_b ; P2 = node @ W2^T  (edge GEMM factored) ----
    k_gemm_bf16<0><<<dim3(EFEAT / 512, N_NODES / 16), T, 0, stream>>>(
        nodeB, W1, proj_b, P1, nullptr, nullptr, N_NODES, EFEAT, FEAT);
    k_gemm_bf16<0><<<dim3(EFEAT / 512, N_NODES / 16), T, 0, stream>>>(
        nodeB, W2, nullptr, P2, nullptr, nullptr, N_NODES, EFEAT, FEAT);

    // ---- heads ----
    k_node_head<<<N_NODES / 8, T, 0, stream>>>(node, xyz_w, xyz_b, wpqr_w, wpqr_b, out);
    k_edge_head<<<N_EDGES / 8, T, 0, stream>>>(eidx, P1, P2, xyzR_w, xyzR_b, wpqRw, wpqRb,
                                               out + (size_t)N_NODES * 6);
}
